// Top2TransformerEncoderLayer_1967095022274
// MI455X (gfx1250) — matrix operand order repaired and checked by reading, not compiled, not run on hardware
//
#include <hip/hip_runtime.h>

typedef unsigned short u16;
typedef __bf16 bf16x16 __attribute__((ext_vector_type(16)));
typedef float  floatx8 __attribute__((ext_vector_type(8)));

// ---- problem constants ----
#define BB   8
#define SS   1024
#define DD   1024
#define HH   16
#define DFFC 4096
#define EE   16
#define TT   (BB * SS)        // 8192
#define CC   ((2 * TT) / EE)  // 1024
#define EPSF 1.1920929e-07f

__device__ __forceinline__ u16 f2bf(float f) {
  unsigned u = __float_as_uint(f);
  u += 0x7fffu + ((u >> 16) & 1u);   // round-to-nearest-even
  return (u16)(u >> 16);
}

union Frag {
  bf16x16 b;
  uint4   q[2];
  u16     s[16];
};

__device__ __forceinline__ floatx8 wmma_bf16(const Frag& a, const Frag& bm, floatx8 c) {
  return __builtin_amdgcn_wmma_f32_16x16x32_bf16(false, a.b, false, bm.b, (short)0, c,
                                                 false, false);
}

// ---------------- elementwise convert f32 -> bf16 ----------------
__global__ void cvt_f32_bf16_kernel(const float* __restrict__ in, u16* __restrict__ out,
                                    long long n) {
  long long i = (long long)blockIdx.x * blockDim.x + threadIdx.x;
  long long stride = (long long)gridDim.x * blockDim.x;
  for (; i < n; i += stride) out[i] = f2bf(in[i]);
}

// ---------------- batched transpose + convert: in[e][R][Cc] -> out[e][Cc][R] ----------------
__global__ void transpose_cvt_kernel(const float* __restrict__ in, u16* __restrict__ out,
                                     int R, int Cc) {
  __shared__ float tile[32][33];
  size_t base = (size_t)blockIdx.z * R * Cc;
  int r0 = blockIdx.y * 32, c0 = blockIdx.x * 32;
  for (int i = threadIdx.y; i < 32; i += 8) {
    tile[i][threadIdx.x] = in[base + (size_t)(r0 + i) * Cc + (c0 + threadIdx.x)];
  }
  __syncthreads();
  for (int i = threadIdx.y; i < 32; i += 8) {
    out[base + (size_t)(c0 + i) * R + (r0 + threadIdx.x)] = f2bf(tile[threadIdx.x][i]);
  }
}

// ---------------- generic bf16 WMMA GEMM: C = A[M,K] * BT[N,K]^T + bias ----------------
template <int OUT_BF16, int RELU>
__global__ __launch_bounds__(256) void gemm_bf16_kernel(
    const u16* __restrict__ A, const u16* __restrict__ BT, const float* __restrict__ bias,
    void* __restrict__ Cout, int M, int N, int K,
    long long sA, long long sB, long long sBias, long long sC) {
  int z = blockIdx.z;
  A  += (size_t)z * sA;
  BT += (size_t)z * sB;
  const float* bi = bias ? bias + (size_t)z * sBias : nullptr;

  int lane = threadIdx.x & 31, wave = threadIdx.x >> 5;
  int wm = wave >> 1, wn = wave & 1;
  int m0 = blockIdx.y * 128 + wm * 32;
  int n0 = blockIdx.x * 128 + wn * 64;
  int l15 = lane & 15, hl = lane >> 4;
  int kh8 = hl * 8, ks16 = hl * 16;

  const u16* a0 = A + (size_t)(m0 + l15) * K;
  const u16* a1 = a0 + (size_t)16 * K;
  const u16* b0 = BT + (size_t)(n0 + l15) * K;

  const floatx8 fz = {0, 0, 0, 0, 0, 0, 0, 0};
  floatx8 acc[2][4];
#pragma unroll
  for (int mi = 0; mi < 2; ++mi)
#pragma unroll
    for (int ni = 0; ni < 4; ++ni) acc[mi][ni] = fz;

  for (int k0 = 0; k0 < K; k0 += 32) {
    Frag a[2], b[4];
    a[0].q[0] = *(const uint4*)(a0 + k0 + kh8);
    a[0].q[1] = *(const uint4*)(a0 + k0 + kh8 + 16);
    a[1].q[0] = *(const uint4*)(a1 + k0 + kh8);
    a[1].q[1] = *(const uint4*)(a1 + k0 + kh8 + 16);
#pragma unroll
    for (int ni = 0; ni < 4; ++ni) {
      const u16* bp = b0 + (size_t)ni * 16 * K + k0 + kh8;
      b[ni].q[0] = *(const uint4*)(bp);
      b[ni].q[1] = *(const uint4*)(bp + 16);
    }
#pragma unroll
    for (int mi = 0; mi < 2; ++mi)
#pragma unroll
      for (int ni = 0; ni < 4; ++ni) acc[mi][ni] = wmma_bf16(a[mi], b[ni], acc[mi][ni]);
  }

#pragma unroll
  for (int ni = 0; ni < 4; ++ni) {
    int col = n0 + ni * 16 + l15;
    float bv = bi ? bi[col] : 0.0f;
#pragma unroll
    for (int mi = 0; mi < 2; ++mi) {
#pragma unroll
      for (int r = 0; r < 8; ++r) {
        int row = m0 + mi * 16 + r + hl * 8;
        float v = acc[mi][ni][r] + bv;
        if (RELU) v = fmaxf(v, 0.0f);
        size_t off = (size_t)z * sC + (size_t)row * N + col;
        if (OUT_BF16) ((u16*)Cout)[off] = f2bf(v);
        else          ((float*)Cout)[off] = v;
      }
    }
  }
}

// ---------------- split heads: qkv f32 [T,3D] -> q,k [BH,S,64] bf16 ; v^T [BH,64,S] bf16 ----------------
__global__ void split_heads_kernel(const float* __restrict__ qkv, u16* __restrict__ q,
                                   u16* __restrict__ k, u16* __restrict__ vt) {
  int stride = gridDim.x * blockDim.x;
  for (int i = blockIdx.x * blockDim.x + threadIdx.x; i < TT * DD; i += stride) {
    int t = i >> 10, d = i & 1023;
    int b = t >> 10, s = t & 1023;
    int h = d >> 6, c = d & 63;
    const float* row = qkv + (size_t)t * (3 * DD);
    int bh = b * HH + h;
    q[((size_t)bh * SS + s) * 64 + c]  = f2bf(row[d]);
    k[((size_t)bh * SS + s) * 64 + c]  = f2bf(row[DD + d]);
    vt[((size_t)bh * 64 + c) * SS + s] = f2bf(row[2 * DD + d]);
  }
}

// ---------------- attention: one wave per (b,h, 16-row q tile) ----------------
__global__ __launch_bounds__(32) void attention_kernel(const u16* __restrict__ qbh,
                                                       const u16* __restrict__ kbh,
                                                       const u16* __restrict__ vt,
                                                       u16* __restrict__ obf) {
  __shared__ float sS[16 * 1024];  // 64 KB scores tile
  int lane = threadIdx.x;
  int nqt = SS / 16;
  int bh = blockIdx.x / nqt, qt = blockIdx.x % nqt;
  int b = bh / HH, h = bh % HH;
  const u16* qb = qbh + (size_t)bh * SS * 64;
  const u16* kb = kbh + (size_t)bh * SS * 64;
  const u16* vb = vt + (size_t)bh * 64 * SS;

  int l15 = lane & 15, hl = lane >> 4;
  int kh8 = hl * 8, ks16 = hl * 16;
  const floatx8 fz = {0, 0, 0, 0, 0, 0, 0, 0};

  // ---- phase 1: scores = q @ k^T * 0.125 ----
  Frag aq[2];
  const u16* ap = qb + (size_t)(qt * 16 + l15) * 64;
  aq[0].q[0] = *(const uint4*)(ap + kh8);
  aq[0].q[1] = *(const uint4*)(ap + kh8 + 16);
  aq[1].q[0] = *(const uint4*)(ap + 32 + kh8);
  aq[1].q[1] = *(const uint4*)(ap + 32 + kh8 + 16);

  for (int nt = 0; nt < SS / 16; ++nt) {
    Frag bk0, bk1;
    const u16* bp = kb + (size_t)(nt * 16 + l15) * 64;
    bk0.q[0] = *(const uint4*)(bp + kh8);
    bk0.q[1] = *(const uint4*)(bp + kh8 + 16);
    bk1.q[0] = *(const uint4*)(bp + 32 + kh8);
    bk1.q[1] = *(const uint4*)(bp + 32 + kh8 + 16);
    floatx8 acc = fz;
    acc = wmma_bf16(aq[0], bk0, acc);
    acc = wmma_bf16(aq[1], bk1, acc);
#pragma unroll
    for (int r = 0; r < 8; ++r)
      sS[(r + hl * 8) * 1024 + nt * 16 + l15] = acc[r] * 0.125f;
  }
  __syncthreads();

  // ---- phase 2: softmax (rows split between lane halves) ----
  float* rp = &sS[l15 * 1024 + hl * 512];
  float mx = -3.4e38f;
  for (int c = 0; c < 512; ++c) mx = fmaxf(mx, rp[c]);
  mx = fmaxf(mx, __shfl_xor(mx, 16, 32));
  float sum = 0.0f;
  for (int c = 0; c < 512; ++c) {
    float e = __expf(rp[c] - mx);
    rp[c] = e;
    sum += e;
  }
  sum += __shfl_xor(sum, 16, 32);
  float rsum = 1.0f / sum;
  __syncthreads();

  // ---- phase 3: out = (P @ V) * rsum ----
  floatx8 acc[4];
#pragma unroll
  for (int nt = 0; nt < 4; ++nt) acc[nt] = fz;
  for (int k0 = 0; k0 < SS; k0 += 32) {
    Frag a2;
    const float* src = &sS[l15 * 1024 + k0 + kh8];
#pragma unroll
    for (int j = 0; j < 8; ++j) {
      a2.s[j] = f2bf(src[j]);
      a2.s[8 + j] = f2bf(src[16 + j]);
    }
    Frag bv[4];
#pragma unroll
    for (int nt = 0; nt < 4; ++nt) {
      const u16* vp = vb + (size_t)(nt * 16 + l15) * SS + k0 + kh8;
      bv[nt].q[0] = *(const uint4*)(vp);
      bv[nt].q[1] = *(const uint4*)(vp + 16);
    }
#pragma unroll
    for (int nt = 0; nt < 4; ++nt) acc[nt] = wmma_bf16(a2, bv[nt], acc[nt]);
  }
  int t0 = b * SS + qt * 16;
#pragma unroll
  for (int nt = 0; nt < 4; ++nt) {
#pragma unroll
    for (int r = 0; r < 8; ++r) {
      int rr = r + hl * 8;
      float scale = __shfl(rsum, rr, 32);
      obf[(size_t)(t0 + rr) * DD + h * 64 + nt * 16 + l15] = f2bf(acc[nt][r] * scale);
    }
  }
}

// ---------------- residual add + layernorm (optional bf16 copy) ----------------
__global__ __launch_bounds__(256) void add_ln_kernel(const float* __restrict__ xin,
                                                     const float* __restrict__ dlt,
                                                     const float* __restrict__ gam,
                                                     const float* __restrict__ bet,
                                                     float* __restrict__ outf,
                                                     u16* __restrict__ outb) {
  __shared__ float red[256];
  int t = blockIdx.x, tid = threadIdx.x;
  const float* xr = xin + (size_t)t * DD;
  const float* dr = dlt + (size_t)t * DD;
  float v[4], s = 0.0f;
#pragma unroll
  for (int i = 0; i < 4; ++i) {
    int d = tid + i * 256;
    v[i] = xr[d] + dr[d];
    s += v[i];
  }
  red[tid] = s;
  __syncthreads();
  for (int off = 128; off; off >>= 1) {
    if (tid < off) red[tid] += red[tid + off];
    __syncthreads();
  }
  float mean = red[0] * (1.0f / DD);
  __syncthreads();
  float s2 = 0.0f;
#pragma unroll
  for (int i = 0; i < 4; ++i) {
    float dv = v[i] - mean;
    s2 += dv * dv;
  }
  red[tid] = s2;
  __syncthreads();
  for (int off = 128; off; off >>= 1) {
    if (tid < off) red[tid] += red[tid + off];
    __syncthreads();
  }
  float rstd = rsqrtf(red[0] * (1.0f / DD) + 1e-5f);
#pragma unroll
  for (int i = 0; i < 4; ++i) {
    int d = tid + i * 256;
    float o = (v[i] - mean) * rstd * gam[d] + bet[d];
    outf[(size_t)t * DD + d] = o;
    if (outb) outb[(size_t)t * DD + d] = f2bf(o);
  }
}

// ---------------- gating: logits, softmax, top-2 ----------------
__global__ __launch_bounds__(256) void gating_kernel(const float* __restrict__ x1f,
                                                     const float* __restrict__ wg,
                                                     int* __restrict__ idx1, int* __restrict__ idx2,
                                                     float* __restrict__ gate1,
                                                     float* __restrict__ gate2) {
  __shared__ float red[256];
  __shared__ float sl[16];
  int t = blockIdx.x, tid = threadIdx.x;
  int e = tid >> 4, j = tid & 15;
  const float* xr = x1f + (size_t)t * DD;
  float p = 0.0f;
  for (int d = j; d < DD; d += 16) p += xr[d] * wg[d * EE + e];
  red[tid] = p;
  __syncthreads();
  for (int off = 8; off; off >>= 1) {
    if (j < off) red[tid] += red[tid + off];
    __syncthreads();
  }
  if (j == 0) sl[e] = red[tid];
  __syncthreads();
  if (tid == 0) {
    float mx = sl[0];
    int a1 = 0;
    for (int q = 1; q < EE; ++q)
      if (sl[q] > mx) { mx = sl[q]; a1 = q; }
    float ex[16], se = 0.0f;
    for (int q = 0; q < EE; ++q) { ex[q] = __expf(sl[q] - mx); se += ex[q]; }
    float mx2 = -3.4e38f;
    int a2 = (a1 == 0) ? 1 : 0;
    for (int q = 0; q < EE; ++q)
      if (q != a1 && sl[q] > mx2) { mx2 = sl[q]; a2 = q; }
    float inv = 1.0f / se;
    idx1[t] = a1; idx2[t] = a2;
    gate1[t] = ex[a1] * inv;
    gate2[t] = ex[a2] * inv;
  }
}

// ---------------- capacity assignment (token-order cumsum semantics) ----------------
__global__ void assign_kernel(const int* __restrict__ idx1, const int* __restrict__ idx2,
                              int* __restrict__ p1, int* __restrict__ p2) {
  int e = threadIdx.x;
  if (e >= EE) return;
  int cnt = 0;
  for (int t = 0; t < TT; ++t)
    if (idx1[t] == e) { p1[t] = (cnt < CC) ? cnt : -1; cnt++; }
  int cnt2 = cnt;  // top-2 positions offset by pre-drop top-1 counts
  for (int t = 0; t < TT; ++t)
    if (idx2[t] == e) { p2[t] = (cnt2 < CC) ? cnt2 : -1; cnt2++; }
}

// ---------------- dispatch tokens into expert buffers ----------------
__global__ __launch_bounds__(256) void dispatch_kernel(const u16* __restrict__ x1b,
                                                       const int* __restrict__ idx1,
                                                       const int* __restrict__ idx2,
                                                       const int* __restrict__ p1,
                                                       const int* __restrict__ p2,
                                                       u16* __restrict__ buf) {
  int t = blockIdx.x, tid = threadIdx.x;
  const uint2* src = (const uint2*)(x1b + (size_t)t * DD);
  uint2 val = src[tid];
  int q1 = p1[t];
  if (q1 >= 0) ((uint2*)(buf + ((size_t)idx1[t] * CC + q1) * DD))[tid] = val;
  int q2 = p2[t];
  if (q2 >= 0) ((uint2*)(buf + ((size_t)idx2[t] * CC + q2) * DD))[tid] = val;
}

// ---------------- combine + final layernorm ----------------
__global__ __launch_bounds__(256) void combine_ln2_kernel(
    const float* __restrict__ x1f, const float* __restrict__ y, const int* __restrict__ idx1,
    const int* __restrict__ idx2, const int* __restrict__ p1, const int* __restrict__ p2,
    const float* __restrict__ gate1, const float* __restrict__ gate2,
    const float* __restrict__ gam, const float* __restrict__ bet, float* __restrict__ out) {
  __shared__ float red[256];
  int t = blockIdx.x, tid = threadIdx.x;
  int q1 = p1[t], q2 = p2[t];
  float g1 = (q1 >= 0) ? gate1[t] : 0.0f;
  float g2 = (q2 >= 0) ? gate2[t] : 0.0f;
  float den = fmaxf(g1 + g2, EPSF);
  g1 /= den; g2 /= den;
  const float* y1 = y + ((size_t)idx1[t] * CC + (q1 >= 0 ? q1 : 0)) * DD;
  const float* y2 = y + ((size_t)idx2[t] * CC + (q2 >= 0 ? q2 : 0)) * DD;
  const float* xr = x1f + (size_t)t * DD;
  float v[4], s = 0.0f;
#pragma unroll
  for (int i = 0; i < 4; ++i) {
    int d = tid + i * 256;
    v[i] = xr[d] + g1 * y1[d] + g2 * y2[d];
    s += v[i];
  }
  red[tid] = s;
  __syncthreads();
  for (int off = 128; off; off >>= 1) {
    if (tid < off) red[tid] += red[tid + off];
    __syncthreads();
  }
  float mean = red[0] * (1.0f / DD);
  __syncthreads();
  float s2 = 0.0f;
#pragma unroll
  for (int i = 0; i < 4; ++i) {
    float dv = v[i] - mean;
    s2 += dv * dv;
  }
  red[tid] = s2;
  __syncthreads();
  for (int off = 128; off; off >>= 1) {
    if (tid < off) red[tid] += red[tid + off];
    __syncthreads();
  }
  float rstd = rsqrtf(red[0] * (1.0f / DD) + 1e-5f);
#pragma unroll
  for (int i = 0; i < 4; ++i) {
    int d = tid + i * 256;
    out[(size_t)t * DD + d] = (v[i] - mean) * rstd * gam[d] + bet[d];
  }
}

// =========================================================================
extern "C" void kernel_launch(void* const* d_in, const int* in_sizes, int n_in,
                              void* d_out, int out_size, void* d_ws, size_t ws_size,
                              hipStream_t stream) {
  (void)in_sizes; (void)n_in; (void)out_size; (void)ws_size;
  const float* x          = (const float*)d_in[0];
  const float* in_proj_w  = (const float*)d_in[1];
  const float* in_proj_b  = (const float*)d_in[2];
  const float* out_proj_w = (const float*)d_in[3];
  const float* out_proj_b = (const float*)d_in[4];
  const float* ln1_g      = (const float*)d_in[5];
  const float* ln1_b      = (const float*)d_in[6];
  const float* ln2_g      = (const float*)d_in[7];
  const float* ln2_b      = (const float*)d_in[8];
  const float* wg         = (const float*)d_in[9];
  const float* w1         = (const float*)d_in[10];
  const float* b1         = (const float*)d_in[11];
  const float* w2         = (const float*)d_in[12];
  const float* b2         = (const float*)d_in[13];
  float* out = (float*)d_out;

  char* ws = (char*)d_ws;
  size_t off = 0;
  auto alloc = [&](size_t bytes) -> char* {
    char* p = ws + off;
    off += (bytes + 255) & ~(size_t)255;
    return p;
  };

  const size_t SZ_XBF  = (size_t)TT * DD * 2;
  const size_t SZ_WIN  = (size_t)3 * DD * DD * 2;
  const size_t SZ_WOUT = (size_t)DD * DD * 2;
  const size_t SZ_W1T  = (size_t)EE * DFFC * DD * 2;
  const size_t SZ_QKV  = (size_t)TT * 3 * DD * 4;
  const size_t SZ_QKH  = (size_t)BB * HH * SS * 64 * 2;
  const size_t SZ_OBF  = (size_t)TT * DD * 2;
  const size_t SZ_F32TD = (size_t)TT * DD * 4;
  const size_t SZ_H    = (size_t)EE * CC * DFFC * 2;
  const size_t SZ_Y    = (size_t)EE * CC * DD * 4;
  const size_t SZ_BUF  = (size_t)EE * CC * DD * 2;
  size_t bigA = SZ_QKV + 3 * SZ_QKH + SZ_OBF;
  size_t bigB = SZ_H + SZ_Y;
  const size_t SZ_BIG = bigA > bigB ? bigA : bigB;

  u16* xbf   = (u16*)alloc(SZ_XBF);
  u16* winb  = (u16*)alloc(SZ_WIN);
  u16* woutb = (u16*)alloc(SZ_WOUT);
  u16* w1t   = (u16*)alloc(SZ_W1T);
  u16* w2t   = (u16*)alloc(SZ_W1T);
  float* oproj = (float*)alloc(SZ_F32TD);
  float* x1f   = (float*)alloc(SZ_F32TD);
  u16* x1b     = (u16*)alloc(SZ_XBF);
  u16* bufE    = (u16*)alloc(SZ_BUF);
  int* idx1    = (int*)alloc(TT * 4);
  int* idx2    = (int*)alloc(TT * 4);
  int* p1      = (int*)alloc(TT * 4);
  int* p2      = (int*)alloc(TT * 4);
  float* gate1 = (float*)alloc(TT * 4);
  float* gate2 = (float*)alloc(TT * 4);
  char* big    = alloc(SZ_BIG);
  // stage-A aliases
  float* qkvf = (float*)big;
  u16* qbh = (u16*)(big + SZ_QKV);
  u16* kbh = (u16*)(big + SZ_QKV + SZ_QKH);
  u16* vtb = (u16*)(big + SZ_QKV + 2 * SZ_QKH);
  u16* obf = (u16*)(big + SZ_QKV + 3 * SZ_QKH);
  // stage-B aliases (reuse after attention/out-proj are done)
  u16* hbuf = (u16*)big;
  float* ybuf = (float*)(big + SZ_H);

  // 1. conversions / weight transposes
  cvt_f32_bf16_kernel<<<4096, 256, 0, stream>>>(x, xbf, (long long)TT * DD);
  cvt_f32_bf16_kernel<<<2048, 256, 0, stream>>>(in_proj_w, winb, (long long)3 * DD * DD);
  cvt_f32_bf16_kernel<<<1024, 256, 0, stream>>>(out_proj_w, woutb, (long long)DD * DD);
  transpose_cvt_kernel<<<dim3(DFFC / 32, DD / 32, EE), dim3(32, 8), 0, stream>>>(w1, w1t, DD, DFFC);
  transpose_cvt_kernel<<<dim3(DD / 32, DFFC / 32, EE), dim3(32, 8), 0, stream>>>(w2, w2t, DFFC, DD);

  // 2. qkv = x @ in_proj_w^T + b   (BT = in_proj_w row-major [3D, D])
  gemm_bf16_kernel<0, 0><<<dim3(3 * DD / 128, TT / 128, 1), 256, 0, stream>>>(
      xbf, winb, in_proj_b, qkvf, TT, 3 * DD, DD, 0, 0, 0, (long long)TT * 3 * DD);

  // 3. split heads + transpose V
  split_heads_kernel<<<16384, 256, 0, stream>>>(qkvf, qbh, kbh, vtb);

  // 4. attention
  attention_kernel<<<BB * HH * (SS / 16), 32, 0, stream>>>(qbh, kbh, vtb, obf);

  // 5. out projection
  gemm_bf16_kernel<0, 0><<<dim3(DD / 128, TT / 128, 1), 256, 0, stream>>>(
      obf, woutb, out_proj_b, oproj, TT, DD, DD, 0, 0, 0, (long long)TT * DD);

  // 6. x1 = LN(x + oproj)
  add_ln_kernel<<<TT, 256, 0, stream>>>(x, oproj, ln1_g, ln1_b, x1f, x1b);

  // 7. gating + top-2 + capacity assignment
  gating_kernel<<<TT, 256, 0, stream>>>(x1f, wg, idx1, idx2, gate1, gate2);
  assign_kernel<<<1, 32, 0, stream>>>(idx1, idx2, p1, p2);

  // 8. dispatch into expert buffers
  hipMemsetAsync(bufE, 0, SZ_BUF, stream);
  dispatch_kernel<<<TT, 256, 0, stream>>>(x1b, idx1, idx2, p1, p2, bufE);

  // 9. expert FFN (batched over E)
  gemm_bf16_kernel<1, 1><<<dim3(DFFC / 128, CC / 128, EE), 256, 0, stream>>>(
      bufE, w1t, b1, hbuf, CC, DFFC, DD,
      (long long)CC * DD, (long long)DFFC * DD, DFFC, (long long)CC * DFFC);
  gemm_bf16_kernel<0, 0><<<dim3(DD / 128, CC / 128, EE), 256, 0, stream>>>(
      hbuf, w2t, b2, ybuf, CC, DD, DFFC,
      (long long)CC * DFFC, (long long)DD * DFFC, DD, (long long)CC * DD);

  // 10. combine + LN2 -> output
  combine_ln2_kernel<<<TT, 256, 0, stream>>>(x1f, ybuf, idx1, idx2, p1, p2, gate1, gate2,
                                             ln2_g, ln2_b, out);
}
